// MultiHeadSelfAttention_31361851195577
// MI455X (gfx1250) — matrix operand order repaired and checked by reading, not compiled, not run on hardware
//
#include <hip/hip_runtime.h>
#include <hip/hip_bf16.h>
#include <stdint.h>

// ---------------------------------------------------------------------------
// MI455X (gfx1250) multi-head self-attention, bf16 WMMA + async-LDS pipeline.
// ---------------------------------------------------------------------------

typedef __attribute__((ext_vector_type(16))) __bf16 bf16x16;
typedef __attribute__((ext_vector_type(8)))  float  f32x8;

union AFrag { bf16x16 v; uint4 q[2]; };
union BFrag { bf16x16 v; uint4 q[2]; };

__device__ __forceinline__ f32x8 wmma_bf16(bf16x16 a, bf16x16 b, f32x8 c) {
  // v_wmma_f32_16x16x32_bf16  D = A*B + C   (wave32, f32 accum)
  return __builtin_amdgcn_wmma_f32_16x16x32_bf16(false, a, false, b,
                                                 (short)0, c, false, false);
}

// Async global -> LDS copy (16B per lane), tracked by ASYNCcnt.
__device__ __forceinline__ void async_b128(void* lds_ptr, const void* gptr) {
  uint32_t lo = (uint32_t)(uintptr_t)lds_ptr;   // low 32b of flat = LDS offset
  asm volatile("global_load_async_to_lds_b128 %0, %1, off"
               :: "v"(lo), "v"(gptr) : "memory");
}
__device__ __forceinline__ void wait_async0() {
  asm volatile("s_wait_asynccnt 0" ::: "memory");
}
__device__ __forceinline__ void wait_async6() {
  asm volatile("s_wait_asynccnt 6" ::: "memory");
}

// ---------------------------------------------------------------------------
// Problem constants
// ---------------------------------------------------------------------------
#define S_LEN   2048
#define D_MODEL 2048
#define N_HEADS 16
#define D_K     128
#define BATCH   2
#define M_TOT   (BATCH * S_LEN)          // 4096
#define LN1E4_OVER_HALF 0.14391156831212787f   // ln(10000)/64
#define ATTN_SCALE      0.08838834764831843f   // 1/sqrt(128)

// ---------------------------------------------------------------------------
// f32 -> bf16 conversion (vectorized, 8 elems/thread)
// ---------------------------------------------------------------------------
__global__ __launch_bounds__(256) void cvt_bf16(const float* __restrict__ src,
                                                __bf16* __restrict__ dst, int n) {
  int i = (blockIdx.x * 256 + threadIdx.x) * 8;
  if (i >= n) return;
  float4 a = *(const float4*)(src + i);
  float4 b = *(const float4*)(src + i + 4);
  union { __bf16 h[8]; uint4 q; } u;
  u.h[0] = (__bf16)a.x; u.h[1] = (__bf16)a.y;
  u.h[2] = (__bf16)a.z; u.h[3] = (__bf16)a.w;
  u.h[4] = (__bf16)b.x; u.h[5] = (__bf16)b.y;
  u.h[6] = (__bf16)b.z; u.h[7] = (__bf16)b.w;
  *(uint4*)(dst + i) = u.q;
}

// ---------------------------------------------------------------------------
// Generic NT GEMM:  C[m,n] = sum_k A[m,k] * Bw[n,k]     (M=4096,N=2048,K=2048)
// Block tile 128x256, 8 waves of 64x64 (16 WMMA : 16 ds_load_b128 per K-step),
// K-step 32, double-buffered async-LDS pipeline (6 async b128 / thread / buf).
//   MODE 0: f32 out (final Wo projection)
//   MODE 1: RoPE epilogue -> bf16 [b,h,s,dk]   (Q and K)
//   MODE 2: bf16 transposed -> [b,h,dk,s]      (V)
// ---------------------------------------------------------------------------
template <int MODE>
__global__ __launch_bounds__(256)
void gemm128(const __bf16* __restrict__ A, const __bf16* __restrict__ Bw,
             float* __restrict__ outF, __bf16* __restrict__ dstB,
             const int* __restrict__ tokpos) {
  const int n0   = blockIdx.x * 256;
  const int m0   = blockIdx.y * 128;
  const int tid  = threadIdx.x;
  const int lane = tid & 31;
  const int w    = tid >> 5;
  const int lr   = lane & 15;
  const int hi   = lane >> 4;
  const int wm   = w >> 2;        // 0..1 -> 64-row half
  const int wn   = w & 3;         // 0..3 -> 64-col quarter

  constexpr int LP = 40;          // LDS row pitch (bf16), 16B-aligned, padded
  __shared__ __align__(16) __bf16 As[2][128 * LP];   // 10 KB / buf
  __shared__ __align__(16) __bf16 Bs[2][256 * LP];   // 20 KB / buf

  const int lrow  = tid >> 1;     // 0..127 (A tile row)
  const int lhalf = tid & 1;      // 0..1   (16 bf16 each)

  auto issue = [&](int it) {
    const int buf = it & 1;
    const int kk  = it * 32;
    // A tile: 128 rows x 32, two threads per row
    const __bf16* ga = A + (size_t)(m0 + lrow) * D_MODEL + kk + lhalf * 16;
    __bf16* la = &As[buf][lrow * LP + lhalf * 16];
    async_b128(la,     ga);
    async_b128(la + 8, ga + 8);
    // B tile: 256 rows x 32, one thread per row (4 x b128)
    const __bf16* gb = Bw + (size_t)(n0 + tid) * D_MODEL + kk;
    __bf16* lb = &Bs[buf][tid * LP];
    async_b128(lb,      gb);
    async_b128(lb + 8,  gb + 8);
    async_b128(lb + 16, gb + 16);
    async_b128(lb + 24, gb + 24);
  };

  f32x8 acc[4][4] = {};

  issue(0);
  const int NIT = D_MODEL / 32;   // 64
  for (int it = 0; it < NIT; ++it) {
    if (it + 1 < NIT) { issue(it + 1); wait_async6(); }
    else              { wait_async0(); }
    __syncthreads();

    const int buf = it & 1;
    AFrag a[4];
#pragma unroll
    for (int mf = 0; mf < 4; ++mf) {
      const int rb = (wm * 64 + mf * 16 + lr) * LP;
      a[mf].q[0] = *(const uint4*)&As[buf][rb + hi * 8];
      a[mf].q[1] = *(const uint4*)&As[buf][rb + 16 + hi * 8];
    }
#pragma unroll
    for (int nf = 0; nf < 4; ++nf) {
      BFrag bb;
      const int rb = (wn * 64 + nf * 16 + lr) * LP;
      bb.q[0] = *(const uint4*)&Bs[buf][rb + hi * 8];
      bb.q[1] = *(const uint4*)&Bs[buf][rb + 16 + hi * 8];
#pragma unroll
      for (int mf = 0; mf < 4; ++mf)
        acc[mf][nf] = wmma_bf16(a[mf].v, bb.v, acc[mf][nf]);
    }
    __syncthreads();
  }

  // ---------------- epilogue ----------------
#pragma unroll
  for (int mf = 0; mf < 4; ++mf) {
#pragma unroll
    for (int nf = 0; nf < 4; ++nf) {
#pragma unroll
      for (int r = 0; r < 8; ++r) {
        const int row = m0 + wm * 64 + mf * 16 + r + 8 * hi;  // 0..4095
        const int col = n0 + wn * 64 + nf * 16 + lr;          // 0..2047
        float v = acc[mf][nf][r];
        if constexpr (MODE == 0) {
          outF[(size_t)row * D_MODEL + col] = v;
        } else {
          const int bat = row >> 11;
          const int s   = row & (S_LEN - 1);
          const int h   = col >> 7;
          const int d   = col & (D_K - 1);
          if constexpr (MODE == 1) {
            const int ri = d >> 1;
            const float pos = (float)tokpos[s];
            const float ang = pos * __expf(-(float)ri * LN1E4_OVER_HALF);
            float sn, cs;
            __sincosf(ang, &sn, &cs);
            const float other = __shfl_xor(v, 1, 32);  // paired feature
            const float res = (d & 1) ? (other * sn + v * cs)
                                      : (v * cs - other * sn);
            dstB[(((size_t)(bat * N_HEADS + h)) * S_LEN + s) * D_K + d] =
                (__bf16)res;
          } else {  // MODE 2: V transposed -> [b,h,dk,s]
            dstB[(((size_t)(bat * N_HEADS + h)) * D_K + d) * S_LEN + s] =
                (__bf16)v;
          }
        }
      }
    }
  }
}

// ---------------------------------------------------------------------------
// Flash attention, one (b,h,64-row q-block) per workgroup. 128 threads/4 waves,
// each wave owns 16 q-rows. Streams causal 64-col K/V blocks via async LDS.
// ---------------------------------------------------------------------------
__global__ __launch_bounds__(128)
void attn_kernel(const __bf16* __restrict__ Qg, const __bf16* __restrict__ Kg,
                 const __bf16* __restrict__ Vtg, __bf16* __restrict__ ctx) {
  const int qb  = blockIdx.x;            // 0..31 (64-row q blocks)
  const int h   = blockIdx.y;            // 0..15
  const int bat = blockIdx.z;            // 0..1
  const int bh  = bat * N_HEADS + h;
  const int tid = threadIdx.x;
  const int lane = tid & 31;
  const int w    = tid >> 5;             // 0..3
  const int lr   = lane & 15;
  const int hi   = lane >> 4;
  const int wr   = w * 16;               // wave's q-row base within block

  __shared__ __align__(16) __bf16 Qs[64 * 128];
  __shared__ __align__(16) __bf16 Ks[64 * 128];
  __shared__ __align__(16) __bf16 Vs[128 * 64];       // transposed: [d][k]
  __shared__ __align__(16) __bf16 Ps[4 * 16 * 64];    // per-wave P tiles

  // Q tile: 64 rows x 128 dk, contiguous 16KB in global
  {
    const __bf16* qsrc = Qg + ((size_t)bh * S_LEN + qb * 64) * D_K;
#pragma unroll
    for (int j = 0; j < 8; ++j)
      async_b128(&Qs[tid * 64 + j * 8], qsrc + tid * 64 + j * 8);
  }

  float m_i[8], l_i[8];
  f32x8 o[8] = {};
#pragma unroll
  for (int r = 0; r < 8; ++r) { m_i[r] = -1e30f; l_i[r] = 0.0f; }

  for (int kb = 0; kb <= qb; ++kb) {
    // K tile (contiguous) and V^T tile (row-strided)
    const __bf16* ksrc = Kg + ((size_t)bh * S_LEN + kb * 64) * D_K;
#pragma unroll
    for (int j = 0; j < 8; ++j)
      async_b128(&Ks[tid * 64 + j * 8], ksrc + tid * 64 + j * 8);
    const __bf16* vsrc = Vtg + ((size_t)bh * D_K + tid) * S_LEN + kb * 64;
#pragma unroll
    for (int j = 0; j < 8; ++j)
      async_b128(&Vs[tid * 64 + j * 8], vsrc + j * 8);
    wait_async0();
    __syncthreads();

    // ---- scores = Q . K^T  (M=16, N=64, K=128) ----
    f32x8 sc[4] = {};
#pragma unroll
    for (int ks = 0; ks < 128; ks += 32) {
      AFrag a;
      a.q[0] = *(const uint4*)&Qs[(wr + lr) * 128 + ks + hi * 8];
      a.q[1] = *(const uint4*)&Qs[(wr + lr) * 128 + ks + 16 + hi * 8];
#pragma unroll
      for (int g = 0; g < 4; ++g) {
        BFrag bb;
        bb.q[0] = *(const uint4*)&Ks[(g * 16 + lr) * 128 + ks + hi * 8];
        bb.q[1] = *(const uint4*)&Ks[(g * 16 + lr) * 128 + ks + 16 + hi * 8];
        sc[g] = wmma_bf16(a.v, bb.v, sc[g]);
      }
    }

    // ---- scale + causal mask ----
#pragma unroll
    for (int g = 0; g < 4; ++g) {
#pragma unroll
      for (int r = 0; r < 8; ++r) {
        float v = sc[g][r] * ATTN_SCALE;
        if (kb == qb) {
          const int kc = kb * 64 + g * 16 + lr;
          const int qr = qb * 64 + wr + r + 8 * hi;
          if (kc > qr) v = -1e30f;
        }
        sc[g][r] = v;
      }
    }

    // ---- online softmax (rows live in 16-lane halves) ----
#pragma unroll
    for (int r = 0; r < 8; ++r) {
      float rm = fmaxf(fmaxf(sc[0][r], sc[1][r]), fmaxf(sc[2][r], sc[3][r]));
#pragma unroll
      for (int off = 1; off < 16; off <<= 1)
        rm = fmaxf(rm, __shfl_xor(rm, off, 32));
      const float mn    = fmaxf(m_i[r], rm);
      const float alpha = __expf(m_i[r] - mn);
      float rs = 0.0f;
#pragma unroll
      for (int g = 0; g < 4; ++g) {
        const float p = __expf(sc[g][r] - mn);
        sc[g][r] = p;
        rs += p;
      }
#pragma unroll
      for (int off = 1; off < 16; off <<= 1) rs += __shfl_xor(rs, off, 32);
      l_i[r] = l_i[r] * alpha + rs;
      m_i[r] = mn;
#pragma unroll
      for (int g = 0; g < 8; ++g) o[g][r] *= alpha;
    }

    // ---- stage P (bf16) in wave-private LDS ----
#pragma unroll
    for (int g = 0; g < 4; ++g)
#pragma unroll
      for (int r = 0; r < 8; ++r)
        Ps[w * 1024 + (r + 8 * hi) * 64 + g * 16 + lr] = (__bf16)sc[g][r];

    // ---- O += P . V  (M=16, N=128, K=64) ----
#pragma unroll
    for (int ks = 0; ks < 64; ks += 32) {
      AFrag a;
      a.q[0] = *(const uint4*)&Ps[w * 1024 + lr * 64 + ks + hi * 8];
      a.q[1] = *(const uint4*)&Ps[w * 1024 + lr * 64 + ks + 16 + hi * 8];
#pragma unroll
      for (int g = 0; g < 8; ++g) {
        BFrag bb;
        bb.q[0] = *(const uint4*)&Vs[(g * 16 + lr) * 64 + ks + hi * 8];
        bb.q[1] = *(const uint4*)&Vs[(g * 16 + lr) * 64 + ks + 16 + hi * 8];
        o[g] = wmma_bf16(a.v, bb.v, o[g]);
      }
    }
    __syncthreads();   // before next block overwrites Ks/Vs
  }

  // ---- normalize + write ctx [b,s,d_model] (bf16) ----
#pragma unroll
  for (int g = 0; g < 8; ++g) {
#pragma unroll
    for (int r = 0; r < 8; ++r) {
      const int srow = qb * 64 + wr + r + 8 * hi;
      const int d    = g * 16 + lr;
      const float v  = o[g][r] / l_i[r];
      ctx[((size_t)bat * S_LEN + srow) * D_MODEL + h * D_K + d] = (__bf16)v;
    }
  }
}

// ---------------------------------------------------------------------------
// Launcher
// ---------------------------------------------------------------------------
extern "C" void kernel_launch(void* const* d_in, const int* in_sizes, int n_in,
                              void* d_out, int out_size, void* d_ws,
                              size_t ws_size, hipStream_t stream) {
  (void)in_sizes; (void)n_in; (void)out_size; (void)ws_size;

  const float* x  = (const float*)d_in[0];
  const float* wq = (const float*)d_in[1];
  const float* wk = (const float*)d_in[2];
  const float* wv = (const float*)d_in[3];
  const float* wo = (const float*)d_in[4];
  const int* tokpos = (const int*)d_in[5];
  float* outF = (float*)d_out;

  char* ws = (char*)d_ws;
  const size_t N_X = (size_t)M_TOT * D_MODEL;      // 8.39M elems
  const size_t N_W = (size_t)D_MODEL * D_MODEL;    // 4.19M elems
  __bf16* xb   = (__bf16*)(ws);
  __bf16* wqb  = (__bf16*)(ws + 16777216);
  __bf16* wkb  = (__bf16*)(ws + 25165824);
  __bf16* wvb  = (__bf16*)(ws + 33554432);
  __bf16* wob  = (__bf16*)(ws + 41943040);
  __bf16* Qb   = (__bf16*)(ws + 50331648);   // [b,h,s,dk]
  __bf16* Kb   = (__bf16*)(ws + 67108864);   // [b,h,s,dk]
  __bf16* Vtb  = (__bf16*)(ws + 83886080);   // [b,h,dk,s]
  __bf16* ctxb = (__bf16*)(ws + 100663296);  // [b,s,d_model]

  // 1) convert to bf16
  cvt_bf16<<<(int)(N_X / 8 / 256), 256, 0, stream>>>(x,  xb,  (int)N_X);
  cvt_bf16<<<(int)(N_W / 8 / 256), 256, 0, stream>>>(wq, wqb, (int)N_W);
  cvt_bf16<<<(int)(N_W / 8 / 256), 256, 0, stream>>>(wk, wkb, (int)N_W);
  cvt_bf16<<<(int)(N_W / 8 / 256), 256, 0, stream>>>(wv, wvb, (int)N_W);
  cvt_bf16<<<(int)(N_W / 8 / 256), 256, 0, stream>>>(wo, wob, (int)N_W);

  // 2) QKV projections (+RoPE / transpose epilogues)
  dim3 gg(D_MODEL / 256, M_TOT / 128);   // (8, 32)
  gemm128<1><<<gg, 256, 0, stream>>>(xb, wqb, nullptr, Qb,  tokpos);
  gemm128<1><<<gg, 256, 0, stream>>>(xb, wkb, nullptr, Kb,  tokpos);
  gemm128<2><<<gg, 256, 0, stream>>>(xb, wvb, nullptr, Vtb, tokpos);

  // 3) causal flash attention
  attn_kernel<<<dim3(S_LEN / 64, N_HEADS, BATCH), 128, 0, stream>>>(Qb, Kb, Vtb,
                                                                    ctxb);

  // 4) output projection (f32 out)
  gemm128<0><<<gg, 256, 0, stream>>>(ctxb, wob, outF, nullptr, nullptr);
}
